// LSTMDecoder_75849122447415
// MI455X (gfx1250) — matrix operand order repaired and checked by reading, not compiled, not run on hardware
//
#include <hip/hip_runtime.h>

// Problem dims
#define B_ 16
#define T_ 64
#define L_ 128
#define H_ 512
#define EMB_ 512
#define VOCAB_ 32000

typedef __attribute__((ext_vector_type(16))) __bf16 v16bf;
typedef __attribute__((ext_vector_type(8)))  float  v8f;
typedef __attribute__((ext_vector_type(4)))  unsigned int u32x4;
typedef __attribute__((ext_vector_type(8)))  int i32x8;
typedef __attribute__((ext_vector_type(4)))  int i32x4;

#if defined(__has_builtin)
#if __has_builtin(__builtin_amdgcn_tensor_load_to_lds) && __has_builtin(__builtin_amdgcn_s_wait_tensorcnt)
#define USE_TDM 1
#endif
#endif
#ifndef USE_TDM
#define USE_TDM 0
#endif

union Frag { v16bf v; uint4 u[2]; };

__device__ __forceinline__ v8f vzero8() {
    v8f z;
#pragma unroll
    for (int j = 0; j < 8; ++j) z[j] = 0.f;
    return z;
}

__device__ __forceinline__ float sigm(float x) { return 1.f / (1.f + __expf(-x)); }

__device__ __forceinline__ v8f wmma_bf16(v16bf a, v16bf b, v8f c) {
    // D = A(16x32 bf16) x B(32x16 bf16) + C(16x16 f32)
    return __builtin_amdgcn_wmma_f32_16x16x32_bf16(false, a, false, b, (short)0, c, false, false);
}

// A fragment: lane (l%16)=row m, half=(l/16): K runs [kb+half*8,+8) and [kb+16+half*8,+8)
__device__ __forceinline__ v16bf loadA(const __bf16* Abase, int K, int kb, int lane) {
    const int m = lane & 15;
    const int half8 = (lane >> 4) << 3;
    const __bf16* p = Abase + (size_t)m * K + kb + half8;
    Frag f;
    f.u[0] = *(const uint4*)(p);
    f.u[1] = *(const uint4*)(p + 16);
    return f.v;
}

// B fragment from row-major weight W (N,K): lane col n = nbase + l%16,
// K runs [kb+half*8,+8) and [kb+16+half*8,+8)  (computes x @ W.T)
__device__ __forceinline__ v16bf loadB(const __bf16* W, int K, int nbase, int kb, int lane) {
    const __bf16* p = W + (size_t)(nbase + (lane & 15)) * K + kb + ((lane >> 4) << 3);
    Frag f;
    f.u[0] = *(const uint4*)(p);
    f.u[1] = *(const uint4*)(p + 16);
    return f.v;
}

// ---------------- prep kernels (fp32 -> bf16, weight fusion, gather) --------

__global__ void k_convert(__bf16* dst, const float* src, int n) {
    for (int i = blockIdx.x * blockDim.x + threadIdx.x; i < n; i += gridDim.x * blockDim.x)
        dst[i] = (__bf16)src[i];
}

// W0cat = [Wih0 (2048x1024) | Whh0 (2048x512)] -> (2048,1536)
__global__ void k_w0cat(__bf16* dst, const float* Wih, const float* Whh, int n) {
    for (int i = blockIdx.x * blockDim.x + threadIdx.x; i < n; i += gridDim.x * blockDim.x) {
        int r = i / 1536, c = i - r * 1536;
        float v = (c < 1024) ? Wih[r * 1024 + c] : Whh[r * 512 + (c - 1024)];
        dst[i] = (__bf16)v;
    }
}

// W1cat = [Wih1 (2048x512) | Whh1 (2048x512)] -> (2048,1024)
__global__ void k_w1cat(__bf16* dst, const float* Wih, const float* Whh, int n) {
    for (int i = blockIdx.x * blockDim.x + threadIdx.x; i < n; i += gridDim.x * blockDim.x) {
        int r = i >> 10, c = i & 1023;
        float v = (c < 512) ? Wih[r * 512 + c] : Whh[r * 512 + (c - 512)];
        dst[i] = (__bf16)v;
    }
}

__global__ void k_bias(float* dst, const float* a, const float* b, int n) {
    for (int i = blockIdx.x * blockDim.x + threadIdx.x; i < n; i += gridDim.x * blockDim.x)
        dst[i] = a[i] + b[i];
}

// embedding gather: dst (B*T, 512) bf16
__global__ void k_embed(__bf16* dst, const float* embW, const int* tokens, int n) {
    for (int i = blockIdx.x * blockDim.x + threadIdx.x; i < n; i += gridDim.x * blockDim.x) {
        int row = i >> 9, k = i & 511;
        int tok = tokens[row];
        dst[i] = (__bf16)embW[(size_t)tok * 512 + k];
    }
}

// ---------------- persistent decoder scan (1 WGP, 32 waves) ----------------
// Wave w owns output columns [16w,16w+16) of every H=512-wide quantity.
// c0/c1 cell state lives in WMMA accumulator fragments for the whole scan.

__global__ __launch_bounds__(1024) void k_scan(
    const __bf16* __restrict__ embx,   // (B*T,512)
    const __bf16* __restrict__ W0,     // (2048,1536)
    const __bf16* __restrict__ W1,     // (2048,1024)
    const float*  __restrict__ b0,     // (2048) bih0+bhh0
    const float*  __restrict__ b1,     // (2048)
    const __bf16* __restrict__ Wq,     // attn_in_W (512,512)
    const __bf16* __restrict__ Wo,     // attn_out_W (512,1024)
    const __bf16* __restrict__ encb,   // (B*L,512)
    const int*    __restrict__ emask,  // (B*L)
    __bf16*       __restrict__ states) // (B*T,512)
{
    __shared__ __bf16 s_in0[16 * 1536]; // [x_t | feed | h0]   48 KB
    __shared__ __bf16 s_in1[16 * 1024]; // [h0 | h1]           32 KB
    __shared__ __bf16 s_att[16 * 1024]; // [ctx | h1]          32 KB
    __shared__ float  s_q[16 * 512];    // query               32 KB
    __shared__ float  s_p[16 * 128];    // scores/probs         8 KB

    const int tid  = threadIdx.x;
    const int lane = tid & 31;
    const int w    = tid >> 5;                 // wave 0..31
    const int col  = (w << 4) + (lane & 15);   // owned column in [0,512)
    const int mhi  = (lane >> 4) << 3;         // row offset: m = j + mhi

    for (int i = tid; i < 16 * 1536; i += 1024) s_in0[i] = (__bf16)0.f;
    for (int i = tid; i < 16 * 1024; i += 1024) { s_in1[i] = (__bf16)0.f; s_att[i] = (__bf16)0.f; }

    // per-lane gate biases (constant across timesteps)
    const float bi0 = b0[col],        bf0 = b0[512 + col];
    const float bg0 = b0[1024 + col], bo0 = b0[1536 + col];
    const float bi1 = b1[col],        bf1 = b1[512 + col];
    const float bg1 = b1[1024 + col], bo1 = b1[1536 + col];

    v8f c0 = vzero8(), c1 = vzero8();

    for (int t = 0; t < T_; ++t) {
        // stage x_t (16x512 bf16) into s_in0 segment 0 — one uint4 per thread
        {
            int e = tid * 8;
            int m = e >> 9, k = e & 511;
            *(uint4*)(s_in0 + m * 1536 + k) =
                *(const uint4*)(embx + (size_t)(m * T_ + t) * 512 + k);
        }
        __syncthreads();

        // ---- LSTM cell 0: gates = [x,feed,h0] @ [Wih0|Whh0].T ----
        v8f ai = vzero8(), af = vzero8(), ag = vzero8(), ao = vzero8();
        for (int kb = 0; kb < 1536; kb += 32) {
            v16bf a = loadA(s_in0, 1536, kb, lane);
            ai = wmma_bf16(a, loadB(W0, 1536, 0    + (w << 4), kb, lane), ai);
            af = wmma_bf16(a, loadB(W0, 1536, 512  + (w << 4), kb, lane), af);
            ag = wmma_bf16(a, loadB(W0, 1536, 1024 + (w << 4), kb, lane), ag);
            ao = wmma_bf16(a, loadB(W0, 1536, 1536 + (w << 4), kb, lane), ao);
        }
        float h0v[8];
#pragma unroll
        for (int j = 0; j < 8; ++j) {
            float iv = sigm(ai[j] + bi0), fv = sigm(af[j] + bf0);
            float gv = tanhf(ag[j] + bg0), ov = sigm(ao[j] + bo0);
            float cn = fv * c0[j] + iv * gv;
            c0[j] = cn;
            h0v[j] = ov * tanhf(cn);
        }
        __syncthreads(); // all reads of s_in0 done
#pragma unroll
        for (int j = 0; j < 8; ++j) {
            int m = j + mhi;
            __bf16 hb = (__bf16)h0v[j];
            s_in0[m * 1536 + 1024 + col] = hb; // h0 for next step's LSTM0
            s_in1[m * 1024 + col]        = hb; // h0 for LSTM1
        }
        __syncthreads();

        // ---- LSTM cell 1: gates = [h0,h1] @ [Wih1|Whh1].T ----
        ai = vzero8(); af = vzero8(); ag = vzero8(); ao = vzero8();
        for (int kb = 0; kb < 1024; kb += 32) {
            v16bf a = loadA(s_in1, 1024, kb, lane);
            ai = wmma_bf16(a, loadB(W1, 1024, 0    + (w << 4), kb, lane), ai);
            af = wmma_bf16(a, loadB(W1, 1024, 512  + (w << 4), kb, lane), af);
            ag = wmma_bf16(a, loadB(W1, 1024, 1024 + (w << 4), kb, lane), ag);
            ao = wmma_bf16(a, loadB(W1, 1024, 1536 + (w << 4), kb, lane), ao);
        }
        float h1v[8];
#pragma unroll
        for (int j = 0; j < 8; ++j) {
            float iv = sigm(ai[j] + bi1), fv = sigm(af[j] + bf1);
            float gv = tanhf(ag[j] + bg1), ov = sigm(ao[j] + bo1);
            float cn = fv * c1[j] + iv * gv;
            c1[j] = cn;
            h1v[j] = ov * tanhf(cn);
        }
        __syncthreads(); // all reads of s_in1 done
#pragma unroll
        for (int j = 0; j < 8; ++j) {
            int m = j + mhi;
            __bf16 hb = (__bf16)h1v[j];
            s_in1[m * 1024 + 512 + col] = hb; // h1 (recurrence)
            s_att[m * 1024 + 512 + col] = hb; // h1 (attn concat)
        }
        __syncthreads();

        // ---- q = h1 @ attn_in_W.T ----
        {
            v8f aq = vzero8();
            for (int kb = 0; kb < 512; kb += 32) {
                v16bf a = loadA(s_in1 + 512, 1024, kb, lane);
                aq = wmma_bf16(a, loadB(Wq, 512, (w << 4), kb, lane), aq);
            }
#pragma unroll
            for (int j = 0; j < 8; ++j) s_q[(j + mhi) * 512 + col] = aq[j];
        }
        __syncthreads();

        // ---- scores[b,l] = enc[b,l,:] . q[b,:]  (+ mask) ----
        for (int idx = tid; idx < B_ * L_; idx += 1024) {
            int b = idx >> 7, l = idx & 127;
            const __bf16* ep = encb + (size_t)(b * L_ + l) * 512;
            float s = 0.f;
            for (int e0 = 0; e0 < 512; e0 += 8) {
                uint4 pk = *(const uint4*)(ep + e0);
                const __bf16* ev = (const __bf16*)&pk;
#pragma unroll
                for (int jj = 0; jj < 8; ++jj) s += (float)ev[jj] * s_q[b * 512 + e0 + jj];
            }
            if (emask[b * L_ + l] == 0) s = -__builtin_inff();
            s_p[idx] = s;
        }
        __syncthreads();

        // ---- softmax per batch row (tiny: 16x128) ----
        if (tid < 16) {
            float mx = -__builtin_inff();
            for (int l = 0; l < L_; ++l) mx = fmaxf(mx, s_p[tid * L_ + l]);
            float sum = 0.f;
            for (int l = 0; l < L_; ++l) {
                float e = __expf(s_p[tid * L_ + l] - mx);
                s_p[tid * L_ + l] = e;
                sum += e;
            }
            float r = 1.f / sum;
            for (int l = 0; l < L_; ++l) s_p[tid * L_ + l] *= r;
        }
        __syncthreads();

        // ---- ctx[b,:] = p[b,:] @ enc[b]  -> s_att segment 0 ----
        {
            int base = tid * 8;
            int b = base >> 9, e0 = base & 511;
            float acc[8];
#pragma unroll
            for (int jj = 0; jj < 8; ++jj) acc[jj] = 0.f;
            const __bf16* ep = encb + (size_t)b * L_ * 512 + e0;
            for (int l = 0; l < L_; ++l) {
                float pv = s_p[b * L_ + l];
                uint4 pk = *(const uint4*)(ep + (size_t)l * 512);
                const __bf16* ev = (const __bf16*)&pk;
#pragma unroll
                for (int jj = 0; jj < 8; ++jj) acc[jj] += pv * (float)ev[jj];
            }
#pragma unroll
            for (int jj = 0; jj < 8; ++jj) s_att[b * 1024 + e0 + jj] = (__bf16)acc[jj];
        }
        __syncthreads();

        // ---- feed = tanh([ctx,h1] @ attn_out_W.T) ----
        {
            v8f av = vzero8();
            for (int kb = 0; kb < 1024; kb += 32) {
                v16bf a = loadA(s_att, 1024, kb, lane);
                av = wmma_bf16(a, loadB(Wo, 1024, (w << 4), kb, lane), av);
            }
#pragma unroll
            for (int j = 0; j < 8; ++j) {
                int m = j + mhi;
                __bf16 fb = (__bf16)tanhf(av[j]);
                s_in0[m * 1536 + 512 + col] = fb;                       // input-feed
                states[(size_t)(m * T_ + t) * 512 + col] = fb;          // scan output
            }
        }
        // next iteration's x-copy + barrier orders the feed writes vs. reads
    }
}

// ---------------- generic K=512 WMMA GEMM: C = act(A @ W.T + bias) ---------
// A (M,512) bf16 row-major, W (N,512) bf16 row-major.
// grid: x = N/128 (8 N-tiles per block, one per wave), y = M/16.
// A tile staged to LDS by the Tensor Data Mover (TDM) when available;
// B stream is double-buffered so its loads overlap the WMMAs.

template <int ACT /*0 none,1 relu*/, int OUTBF /*1 bf16 out, 0 f32 out*/>
__global__ __launch_bounds__(256) void k_gemm512(
    const __bf16* __restrict__ A,
    const __bf16* __restrict__ W,
    const float*  __restrict__ bias, // may be null
    float*        __restrict__ Cf,
    __bf16*       __restrict__ Cb,
    int N)
{
    __shared__ __bf16 sA[16 * 512]; // 16 KB staged A tile

    const int tid = threadIdx.x, lane = tid & 31, w = tid >> 5;
    const int mtile = blockIdx.y;
    const int nbase = (blockIdx.x * 8 + w) * 16;

#if USE_TDM
    // One TDM descriptor: contiguous 1-D tile of 8192 bf16 (16 rows x 512 cols).
    if (tid < 32) { // wave 0 issues the DMA (EXEC ignored by tensor ops)
        unsigned lds_off = (unsigned)(size_t)(__bf16*)sA;
        unsigned long long ga =
            (unsigned long long)(size_t)(A + (size_t)mtile * 16 * 512);
        u32x4 g0;
        g0[0] = 1u;                                   // count=1 (valid user D#)
        g0[1] = lds_off;                              // lds_addr (bytes)
        g0[2] = (unsigned)(ga & 0xffffffffu);         // global_addr[31:0]
        g0[3] = (unsigned)((ga >> 32) & 0x1ffffffu)   // global_addr[56:32]
              | (2u << 30);                           // type=2 ("image")
        i32x8 g1;
        g1[0] = (int)(1u << 16);        // workgroup_mask=0, data_size=1 (2B)
        g1[1] = (int)(8192u << 16);     // tensor_dim0[15:0]=8192 in bits[79:64]... [31:16] of dword1
        g1[2] = (int)(1u << 16);        // tensor_dim0 hi=0 | tensor_dim1[15:0]=1
        g1[3] = (int)(8192u << 16);     // tensor_dim1 hi=0 | tile_dim0=8192
        g1[4] = 1;                      // tile_dim1=1, tile_dim2=0
        g1[5] = 8192;                   // tensor_dim0_stride[31:0]
        g1[6] = 0;                      // stride0 hi | stride1 lo
        g1[7] = 0;                      // stride1 hi
        i32x4 g2, g3;
        g2[0] = g2[1] = g2[2] = g2[3] = 0;
        g3[0] = g3[1] = g3[2] = g3[3] = 0;
#if defined(__clang_major__) && (__clang_major__ >= 23)
        i32x8 gx;
        gx[0] = gx[1] = gx[2] = gx[3] = gx[4] = gx[5] = gx[6] = gx[7] = 0;
        __builtin_amdgcn_tensor_load_to_lds(g0, g1, g2, g3, gx, 0);
#else
        __builtin_amdgcn_tensor_load_to_lds(g0, g1, g2, g3, 0);
#endif
        __builtin_amdgcn_s_wait_tensorcnt(0);
    }
    __syncthreads();
#else
    for (int i = tid * 8; i < 16 * 512; i += 256 * 8)
        *(uint4*)(sA + i) = *(const uint4*)(A + (size_t)mtile * 16 * 512 + i);
    __syncthreads();
#endif

    // software-pipelined K loop: B fragment for k+1 in flight during WMMA k
    v8f acc = vzero8();
    v16bf bC = loadB(W, 512, nbase, 0, lane);
    for (int kb = 32; kb < 512; kb += 32) {
        __builtin_prefetch(W + (size_t)(nbase + (lane & 15)) * 512 + kb + 32, 0, 0);
        v16bf aC = loadA(sA, 512, kb - 32, lane);
        v16bf bN = loadB(W, 512, nbase, kb, lane);
        acc = wmma_bf16(aC, bC, acc);
        bC = bN;
    }
    acc = wmma_bf16(loadA(sA, 512, 512 - 32, lane), bC, acc);

    const int n = nbase + (lane & 15);
    const float bv = bias ? bias[n] : 0.f;
#pragma unroll
    for (int j = 0; j < 8; ++j) {
        float v = acc[j] + bv;
        if (ACT == 1) v = fmaxf(v, 0.f);
        int m = mtile * 16 + j + ((lane >> 4) << 3);
        size_t off = (size_t)m * N + n;
        if (OUTBF) Cb[off] = (__bf16)v;
        else       Cf[off] = v;
    }
}

// ---------------------------------------------------------------------------

extern "C" void kernel_launch(void* const* d_in, const int* in_sizes, int n_in,
                              void* d_out, int out_size, void* d_ws, size_t ws_size,
                              hipStream_t stream) {
    const int*   tokens   = (const int*)d_in[0];
    const float* enc      = (const float*)d_in[1];
    const int*   emask    = (const int*)d_in[2];
    const float* embW     = (const float*)d_in[3];
    const float* Wih0     = (const float*)d_in[4];
    const float* Whh0     = (const float*)d_in[5];
    const float* bih0     = (const float*)d_in[6];
    const float* bhh0     = (const float*)d_in[7];
    const float* Wih1     = (const float*)d_in[8];
    const float* Whh1     = (const float*)d_in[9];
    const float* bih1     = (const float*)d_in[10];
    const float* bhh1     = (const float*)d_in[11];
    const float* attnInW  = (const float*)d_in[12];
    const float* attnOutW = (const float*)d_in[13];
    const float* ffnW1    = (const float*)d_in[14];
    const float* ffnb1    = (const float*)d_in[15];
    const float* ffnW2    = (const float*)d_in[16];
    const float* ffnb2    = (const float*)d_in[17];
    (void)in_sizes; (void)n_in; (void)out_size; (void)ws_size;

    char* ws = (char*)d_ws;
    size_t off = 0;
    auto take = [&](size_t bytes) {
        char* p = ws + off;
        off += (bytes + 255) & ~(size_t)255;
        return p;
    };
    __bf16* embx   = (__bf16*)take((size_t)B_ * T_ * EMB_ * 2);
    __bf16* W0c    = (__bf16*)take((size_t)2048 * 1536 * 2);
    __bf16* W1c    = (__bf16*)take((size_t)2048 * 1024 * 2);
    __bf16* Wq_b   = (__bf16*)take((size_t)512 * 512 * 2);
    __bf16* Wo_b   = (__bf16*)take((size_t)512 * 1024 * 2);
    __bf16* enc_b  = (__bf16*)take((size_t)B_ * L_ * H_ * 2);
    __bf16* fW1_b  = (__bf16*)take((size_t)512 * 512 * 2);
    __bf16* fW2_b  = (__bf16*)take((size_t)512 * 512 * 2);
    __bf16* eW_b   = (__bf16*)take((size_t)VOCAB_ * EMB_ * 2);
    float*  b0     = (float*)take((size_t)2048 * 4);
    float*  b1     = (float*)take((size_t)2048 * 4);
    __bf16* states = (__bf16*)take((size_t)B_ * T_ * H_ * 2);
    __bf16* f1     = (__bf16*)take((size_t)B_ * T_ * H_ * 2);
    __bf16* f2     = (__bf16*)take((size_t)B_ * T_ * EMB_ * 2);

    auto grid1 = [](int n) { return dim3((unsigned)((n + 2047) / 2048)); };

    int n;
    n = 2048 * 1536;      k_w0cat  <<<grid1(n), 256, 0, stream>>>(W0c, Wih0, Whh0, n);
    n = 2048 * 1024;      k_w1cat  <<<grid1(n), 256, 0, stream>>>(W1c, Wih1, Whh1, n);
    n = 512 * 512;        k_convert<<<grid1(n), 256, 0, stream>>>(Wq_b,  attnInW,  n);
    n = 512 * 1024;       k_convert<<<grid1(n), 256, 0, stream>>>(Wo_b,  attnOutW, n);
    n = B_ * L_ * H_;     k_convert<<<grid1(n), 256, 0, stream>>>(enc_b, enc,      n);
    n = 512 * 512;        k_convert<<<grid1(n), 256, 0, stream>>>(fW1_b, ffnW1,    n);
    n = 512 * 512;        k_convert<<<grid1(n), 256, 0, stream>>>(fW2_b, ffnW2,    n);
    n = VOCAB_ * EMB_;    k_convert<<<grid1(n), 256, 0, stream>>>(eW_b,  embW,     n);
    n = 2048;             k_bias   <<<grid1(n), 256, 0, stream>>>(b0, bih0, bhh0,  n);
    n = 2048;             k_bias   <<<grid1(n), 256, 0, stream>>>(b1, bih1, bhh1,  n);
    n = B_ * T_ * EMB_;   k_embed  <<<grid1(n), 256, 0, stream>>>(embx, embW, tokens, n);

    // sequential decoder scan: one persistent workgroup (32 wave32s on a WGP)
    k_scan<<<1, 1024, 0, stream>>>(embx, W0c, W1c, b0, b1, Wq_b, Wo_b, enc_b, emask, states);

    // FFN1: relu(states @ ffn_W1.T + b1)  (M=1024, N=512)
    k_gemm512<1, 1><<<dim3(512 / 128, (B_ * T_) / 16), 256, 0, stream>>>(
        states, fW1_b, ffnb1, (float*)nullptr, f1, 512);
    // FFN2: f1 @ ffn_W2.T + b2            (M=1024, N=512)
    k_gemm512<0, 1><<<dim3(512 / 128, (B_ * T_) / 16), 256, 0, stream>>>(
        f1, fW2_b, ffnb2, (float*)nullptr, f2, 512);
    // Vocab projection: f2 @ embed_W.T    (M=1024, N=32000) -> fp32 d_out
    k_gemm512<0, 0><<<dim3(VOCAB_ / 128, (B_ * T_) / 16), 256, 0, stream>>>(
        f2, eW_b, (const float*)nullptr, (float*)d_out, (__bf16*)nullptr, VOCAB_);
}
